// IFESAGE_57921928954040
// MI455X (gfx1250) — matrix operand order repaired and checked by reading, not compiled, not run on hardware
//
#include <hip/hip_runtime.h>

// GraphSAGE x3 + classifier for MI455X (gfx1250, wave32, WMMA).
// Dense GEMMs: v_wmma_f32_16x16x32_bf16 (f32 accumulate), 2 column tiles per
// wave to amortize A-fragment loads/conversions. Graph aggregation: native
// global_atomic_add_f32 (inline asm, device scope) — L2-resident working set.

typedef __bf16 bf16;
typedef __attribute__((ext_vector_type(16))) __bf16 v16bf;
typedef __attribute__((ext_vector_type(8)))  __bf16 bf16x8;
typedef __attribute__((ext_vector_type(8)))  float  v8f;

#define FEAT 128
#define NCLS 40
#define NCLS_PAD 48

// Guaranteed-native f32 atomic add (no CAS fallback). Atomic-without-return is
// tracked by STOREcnt; S_ENDPGM's implicit wait-idle covers completion.
__device__ __forceinline__ void atom_add_f32(float* p, float v) {
  asm volatile("global_atomic_add_f32 %0, %1, off scope:SCOPE_DEV"
               :
               : "v"(p), "v"(v)
               : "memory");
}

// ---------------------------------------------------------------- utility
__global__ void k_zero(float* __restrict__ p, int n) {
  int i = blockIdx.x * blockDim.x + threadIdx.x;
  if (i < n) p[i] = 0.0f;
}

__global__ void k_deg(const int* __restrict__ dst, float* __restrict__ deg, int E) {
  int e = blockIdx.x * blockDim.x + threadIdx.x;
  if (e < E) atom_add_f32(&deg[dst[e]], 1.0f);
}

__global__ void k_deginv(const float* __restrict__ deg, float* __restrict__ dinv, int n) {
  int i = blockIdx.x * blockDim.x + threadIdx.x;
  if (i < n) dinv[i] = 1.0f / fmaxf(deg[i], 1.0f);
}

__global__ void k_f32_to_bf16(const float* __restrict__ in, bf16* __restrict__ out, int n) {
  int i = blockIdx.x * blockDim.x + threadIdx.x;
  if (i < n) out[i] = (bf16)in[i];
}

// Zero-pad the classifier weight [40x128] -> [48x128] so WMMA tiles stay full
// (EXEC must be all-ones through WMMA).
__global__ void k_wc_pad(const float* __restrict__ Wc, bf16* __restrict__ out) {
  int i = blockIdx.x * blockDim.x + threadIdx.x;
  if (i < NCLS_PAD * FEAT) {
    int o = i >> 7, k = i & 127;
    out[i] = (bf16)((o < NCLS) ? Wc[o * FEAT + k] : 0.0f);
  }
}

// ------------------------------------------------------------- scatter-add
// One thread per (edge, float4-chunk): gather x[src], atomic-add into agg[dst].
__global__ void k_scatter(const float* __restrict__ x, const int* __restrict__ src,
                          const int* __restrict__ dst, float* __restrict__ agg, int E) {
  int gid = blockIdx.x * blockDim.x + threadIdx.x;
  if (gid >= E * 32) return;
  int e = gid >> 5, q = gid & 31;
  int s = src[e], d = dst[e];
  float4 v = ((const float4*)(x + (size_t)s * FEAT))[q];
  float* ap = agg + (size_t)d * FEAT + q * 4;
  atom_add_f32(ap + 0, v.x);
  atom_add_f32(ap + 1, v.y);
  atom_add_f32(ap + 2, v.z);
  atom_add_f32(ap + 3, v.w);
}

// --------------------------------------------------------- fragment packing
// 16-bit A 16x32, contiguous K (same order as B): lane half h (=lane>>4), row m (=lane&15):
//   a[i]   = A[m][kb + 16h + i]       i in 0..15
// 16-bit B 32x16 (col o = lane&15): b[i] = B[kb + 16h + i][o] = W[o][kb+16h+i]
// (contiguous K in the row-major weight -> b128 loads).
__device__ __forceinline__ void putf4(v16bf& v, int base, float4 f, float s) {
  v[base + 0] = (bf16)(f.x * s);
  v[base + 1] = (bf16)(f.y * s);
  v[base + 2] = (bf16)(f.z * s);
  v[base + 3] = (bf16)(f.w * s);
}
__device__ __forceinline__ void putb8(v16bf& v, int base, bf16x8 u) {
#pragma unroll
  for (int j = 0; j < 8; ++j) v[base + j] = u[j];
}

// ------------------------------------------------------------- SAGE GEMM
// out = relu( (agg*dinv) @ Wl^T + bias + x @ Wr^T ) [+ residual]
// Block = 128 threads = 4 waves; each wave owns TWO 16-col tiles (col tiles
// 2w and 2w+1) so each A fragment feeds 2 WMMAs.
__global__ __launch_bounds__(128)
void k_sage_gemm(const float* __restrict__ aggF, const float* __restrict__ xF,
                 const float* __restrict__ dinv,
                 const bf16* __restrict__ Wl, const bf16* __restrict__ Wr,
                 const float* __restrict__ bias, const float* __restrict__ residual,
                 float* __restrict__ out, int N) {
  const int lane = threadIdx.x & 31;
  const int wave = threadIdx.x >> 5;  // 0..3
  const int row0 = blockIdx.x << 4;
  const int m    = lane & 15;
  const int half = lane >> 4;
  int row = row0 + m;
  if (row >= N) row = N - 1;  // clamp loads; EXEC stays uniform for WMMA
  const int o0 = (wave * 2) * 16 + m;
  const int o1 = o0 + 16;

  const float s = dinv[row];
  const float4* aP  = (const float4*)(aggF + (size_t)row * FEAT);
  const float4* xP  = (const float4*)(xF   + (size_t)row * FEAT);
  const bf16x8* l0P = (const bf16x8*)(Wl + (size_t)o0 * FEAT);
  const bf16x8* r0P = (const bf16x8*)(Wr + (size_t)o0 * FEAT);
  const bf16x8* l1P = (const bf16x8*)(Wl + (size_t)o1 * FEAT);
  const bf16x8* r1P = (const bf16x8*)(Wr + (size_t)o1 * FEAT);

  v8f c0 = {}, c1 = {};
#pragma unroll
  for (int kc = 0; kc < 4; ++kc) {
    const int e0 = kc * 8 + half * 4;  // float4 index of (32*kc + 16*half)
    const int s0 = kc * 4 + half * 2;  // bf16x8 index of (32*kc + 16*half)
    v16bf aA, aX, b0, b1;
    putf4(aA, 0,  aP[e0],     s);    putf4(aA, 4,  aP[e0 + 1], s);
    putf4(aA, 8,  aP[e0 + 2], s);    putf4(aA, 12, aP[e0 + 3], s);
    putf4(aX, 0,  xP[e0],     1.0f); putf4(aX, 4,  xP[e0 + 1], 1.0f);
    putf4(aX, 8,  xP[e0 + 2], 1.0f); putf4(aX, 12, xP[e0 + 3], 1.0f);
    // Wl path: mean-aggregated neighbors
    putb8(b0, 0, l0P[s0]); putb8(b0, 8, l0P[s0 + 1]);
    putb8(b1, 0, l1P[s0]); putb8(b1, 8, l1P[s0 + 1]);
    c0 = __builtin_amdgcn_wmma_f32_16x16x32_bf16(false, aA, false, b0, (short)0, c0,
                                                 false, false);
    c1 = __builtin_amdgcn_wmma_f32_16x16x32_bf16(false, aA, false, b1, (short)0, c1,
                                                 false, false);
    // Wr path: root features
    putb8(b0, 0, r0P[s0]); putb8(b0, 8, r0P[s0 + 1]);
    putb8(b1, 0, r1P[s0]); putb8(b1, 8, r1P[s0 + 1]);
    c0 = __builtin_amdgcn_wmma_f32_16x16x32_bf16(false, aX, false, b0, (short)0, c0,
                                                 false, false);
    c1 = __builtin_amdgcn_wmma_f32_16x16x32_bf16(false, aX, false, b1, (short)0, c1,
                                                 false, false);
  }

  // C/D layout: c[r] -> (M = 8*half + r, N = lane&15)
  const int colA = (wave * 2) * 16 + m;
  const int colB = colA + 16;
  const float ba = bias[colA];
  const float bb = bias[colB];
#pragma unroll
  for (int r = 0; r < 8; ++r) {
    int rr = row0 + half * 8 + r;
    if (rr < N) {
      float va = c0[r] + ba;
      float vb = c1[r] + bb;
      va = va > 0.0f ? va : 0.0f;
      vb = vb > 0.0f ? vb : 0.0f;
      if (residual) {
        va += residual[(size_t)rr * FEAT + colA];
        vb += residual[(size_t)rr * FEAT + colB];
      }
      out[(size_t)rr * FEAT + colA] = va;
      out[(size_t)rr * FEAT + colB] = vb;
    }
  }
}

// ---------------------------------------------------------- classifier GEMM
// out[N x 40] = h @ Wc^T + bc  (Wc padded to 48 cols; stores guarded to 40)
__global__ __launch_bounds__(96)
void k_cls_gemm(const float* __restrict__ h, const bf16* __restrict__ Wc,
                const float* __restrict__ bc, float* __restrict__ out, int N) {
  const int lane = threadIdx.x & 31;
  const int wave = threadIdx.x >> 5;  // 0..2 -> col tiles 0,16,32
  const int row0 = blockIdx.x << 4;
  const int col0 = wave << 4;
  const int m    = lane & 15;
  const int half = lane >> 4;
  int row = row0 + m;
  if (row >= N) row = N - 1;
  const int o = col0 + m;  // < 48 always (padded weight)

  const float4* hP = (const float4*)(h + (size_t)row * FEAT);
  const bf16x8* wP = (const bf16x8*)(Wc + (size_t)o * FEAT);

  v8f c = {};
#pragma unroll
  for (int kc = 0; kc < 4; ++kc) {
    const int e0 = kc * 8 + half * 4;
    const int s0 = kc * 4 + half * 2;
    v16bf a, b;
    putf4(a, 0,  hP[e0],     1.0f); putf4(a, 4,  hP[e0 + 1], 1.0f);
    putf4(a, 8,  hP[e0 + 2], 1.0f); putf4(a, 12, hP[e0 + 3], 1.0f);
    putb8(b, 0, wP[s0]); putb8(b, 8, wP[s0 + 1]);
    c = __builtin_amdgcn_wmma_f32_16x16x32_bf16(false, a, false, b, (short)0, c,
                                                false, false);
  }
  const int col = col0 + m;
  if (col < NCLS) {  // divergence only AFTER all WMMA
    const float bb = bc[col];
#pragma unroll
    for (int r = 0; r < 8; ++r) {
      int rr = row0 + half * 8 + r;
      if (rr < N) out[(size_t)rr * NCLS + col] = c[r] + bb;
    }
  }
}

// ------------------------------------------------------------------ launch
extern "C" void kernel_launch(void* const* d_in, const int* in_sizes, int n_in,
                              void* d_out, int out_size, void* d_ws, size_t ws_size,
                              hipStream_t stream) {
  const float* x     = (const float*)d_in[0];
  const int*   edges = (const int*)d_in[1];
  const int N = in_sizes[0] / FEAT;
  const int E = in_sizes[1] / 2;
  const int* src = edges;
  const int* dst = edges + E;
  const float* Wl[3] = {(const float*)d_in[2], (const float*)d_in[5], (const float*)d_in[8]};
  const float* bl[3] = {(const float*)d_in[3], (const float*)d_in[6], (const float*)d_in[9]};
  const float* Wr[3] = {(const float*)d_in[4], (const float*)d_in[7], (const float*)d_in[10]};
  const float* Wc = (const float*)d_in[11];
  const float* bc = (const float*)d_in[12];

  // workspace layout (~77 MB): deg, deginv, agg, two activation buffers, bf16 weights
  char* p = (char*)d_ws;
  auto take = [&](size_t bytes) {
    char* r = p;
    p += (bytes + 255) & ~(size_t)255;
    return r;
  };
  float* deg  = (float*)take((size_t)N * 4);
  float* dinv = (float*)take((size_t)N * 4);
  float* agg  = (float*)take((size_t)N * FEAT * 4);
  float* bufA = (float*)take((size_t)N * FEAT * 4);
  float* bufB = (float*)take((size_t)N * FEAT * 4);
  bf16* Wlb[3];
  bf16* Wrb[3];
  for (int i = 0; i < 3; ++i) {
    Wlb[i] = (bf16*)take((size_t)FEAT * FEAT * 2);
    Wrb[i] = (bf16*)take((size_t)FEAT * FEAT * 2);
  }
  bf16* Wcb = (bf16*)take((size_t)NCLS_PAD * FEAT * 2);

  const int B = 256;
  // degrees
  k_zero<<<(N + B - 1) / B, B, 0, stream>>>(deg, N);
  k_deg<<<(E + B - 1) / B, B, 0, stream>>>(dst, deg, E);
  k_deginv<<<(N + B - 1) / B, B, 0, stream>>>(deg, dinv, N);
  // weights -> bf16 (converted once, reused by 3125 row tiles)
  for (int i = 0; i < 3; ++i) {
    k_f32_to_bf16<<<(FEAT * FEAT + B - 1) / B, B, 0, stream>>>(Wl[i], Wlb[i], FEAT * FEAT);
    k_f32_to_bf16<<<(FEAT * FEAT + B - 1) / B, B, 0, stream>>>(Wr[i], Wrb[i], FEAT * FEAT);
  }
  k_wc_pad<<<(NCLS_PAD * FEAT + B - 1) / B, B, 0, stream>>>(Wc, Wcb);

  const int gemmGrid = (N + 15) / 16;
  const float* in = x;
  float* outs[3]      = {bufA, bufB, bufA};
  const float* res[3] = {nullptr, nullptr, bufB};  // layer 3: + global_repr
  for (int L = 0; L < 3; ++L) {
    k_zero<<<((N * FEAT) + B - 1) / B, B, 0, stream>>>(agg, N * FEAT);
    k_scatter<<<((E * 32) + B - 1) / B, B, 0, stream>>>(in, src, dst, agg, E);
    k_sage_gemm<<<gemmGrid, 128, 0, stream>>>(agg, in, dinv, Wlb[L], Wrb[L], bl[L],
                                              res[L], outs[L], N);
    in = outs[L];
  }
  k_cls_gemm<<<gemmGrid, 96, 0, stream>>>(bufA, Wcb, bc, (float*)d_out, N);
}
